// HeteroGNN_36361193128372
// MI455X (gfx1250) — matrix operand order repaired and checked by reading, not compiled, not run on hardware
//
#include <hip/hip_runtime.h>

typedef __attribute__((ext_vector_type(16))) __bf16 v16bf;
typedef __attribute__((ext_vector_type(8)))  float  v8f;

#define H 128
#define OUTC 64
#define NE 300000

// node types: 0 course,1 field,2 resource,3 teacher,4 school,5 user,6 comment,7 reply,8 exercise,9 video
__device__ __constant__ int c_dstRelCnt[10] = {6,1,3,2,3,4,3,2,1,1};
__device__ __constant__ int c_dstRels[10][6] = {
    {1,3,5,7,9,11},      // course
    {0,0,0,0,0,0},       // field
    {2,23,25,0,0,0},     // resource
    {4,20,0,0,0,0},      // teacher
    {6,19,21,0,0,0},     // school
    {8,15,17,18,0,0},    // user
    {10,13,14,0,0,0},    // comment
    {12,16,0,0,0,0},     // reply
    {22,0,0,0,0,0},      // exercise
    {24,0,0,0,0,0}       // video
};

// WMMA B-fragment swizzle for a [128 x nCols] K-major weight matrix:
// fragment element j of (colTile, kChunk, lane) stored contiguously so each lane
// loads its whole 16 x bf16 fragment as one 32-byte vector.
__device__ __forceinline__ size_t bswz(int K, int n, int nCols) {
    int colTile = n >> 4, colN = n & 15;
    int kc = K >> 5, rem = K & 31;
    int lane = (((rem >> 3) & 1) << 4) | colN;
    int j = (rem & 7) | ((rem >> 4) << 3);
    return ((size_t)(colTile * 4 + kc) * 32 + lane) * 16 + j;
}

// ---------- prep: convert f32 weights -> bf16 in swizzled B-fragment layout ----------
__global__ void k_cvt_swz(const float* __restrict__ in, __bf16* __restrict__ out,
                          int nMats, int nCols) {
    int total = nMats * H * nCols;
    int i = blockIdx.x * blockDim.x + threadIdx.x;
    if (i >= total) return;
    int mat = i / (H * nCols);
    int rem = i % (H * nCols);
    int K = rem / nCols;
    int n = rem % nCols;
    out[(size_t)mat * H * nCols + bswz(K, n, nCols)] = (__bf16)in[i];
}

__global__ void k_wrsum(const float* __restrict__ Wr, __bf16* __restrict__ wrsum,
                        const float* __restrict__ bl, float* __restrict__ bsum) {
    int i = blockIdx.x * blockDim.x + threadIdx.x;
    if (i >= 2 * 10 * H * H) return;
    int l   = i / (10 * H * H);
    int rem = i % (10 * H * H);
    int t   = rem / (H * H);
    int e   = rem % (H * H);
    int K   = e / H, n = e % H;
    int cnt = c_dstRelCnt[t];
    float s = 0.0f;
    for (int k = 0; k < cnt; ++k) {
        int r = c_dstRels[t][k];
        s += Wr[((size_t)l * 26 + r) * (H * H) + e];
    }
    wrsum[(size_t)(l * 10 + t) * H * H + bswz(K, n, H)] = (__bf16)s;
    if (e < H) {
        float bs = 0.0f;
        for (int k = 0; k < cnt; ++k)
            bs += bl[((size_t)l * 26 + c_dstRels[t][k]) * H + e];
        bsum[(l * 10 + t) * H + e] = bs;
    }
}

// ---------- edge scatter: agg[dst] += x[src], deg[dst] += 1 ----------
__global__ void k_scatter(const float* __restrict__ X, const int* __restrict__ srcI,
                          const int* __restrict__ dstI, float* __restrict__ agg,
                          float* __restrict__ deg, int nE) {
    int edge = blockIdx.x * (blockDim.x >> 5) + (threadIdx.x >> 5);
    if (edge >= nE) return;
    int lane = threadIdx.x & 31;
    int s = srcI[edge];
    int d = dstI[edge];
    const float4* xs = (const float4*)(X + (size_t)s * H);
    float4 v = xs[lane];                       // 32 lanes x 4 floats = 128
    float* ad = agg + (size_t)d * H + lane * 4;
    atomicAdd(ad + 0, v.x);
    atomicAdd(ad + 1, v.y);
    atomicAdd(ad + 2, v.z);
    atomicAdd(ad + 3, v.w);
    if (lane == 0) atomicAdd(deg + d, 1.0f);
}

// ---------- WMMA GEMM: Out = (A * inv(deg)) @ Bsw (+bias | accumulates into Out) ----------
// blockDim.x = 2*nCols (one wave per 16-col tile). K=128 fixed. A row-major f32 (stride 128),
// Bsw bf16 pre-swizzled fragments, Out row-major f32 [nRows x nCols].
__global__ void k_gemm(const float* __restrict__ A, const float* __restrict__ deg,
                       const __bf16* __restrict__ Bsw, const float* __restrict__ bias,
                       float* __restrict__ Out, int nRows, int nCols, int accumulate) {
    int lane = threadIdx.x & 31;
    int wv   = threadIdx.x >> 5;
    int rowT = blockIdx.x * 16;
    int grow = rowT + (lane & 15);
    int growC = grow < nRows ? grow : nRows - 1;   // clamp: rows >= nRows never stored
    int hi8  = (lane >> 4) * 8;
    int kmod = hi8;                                 // A K-offset for this lane half
    int ocol = wv * 16 + (lane & 15);

    float inv = 1.0f;
    if (deg) inv = 1.0f / fmaxf(deg[growC], 1.0f);

    const float4* A4 = (const float4*)(A + (size_t)growC * H);
    const v16bf*  Bv = (const v16bf*)Bsw;

    v8f c = {};
    if (accumulate) {
#pragma unroll
        for (int v = 0; v < 8; ++v) {
            int om = rowT + hi8 + v;
            if (om >= nRows) om = nRows - 1;        // clamped load; garbage rows unstored
            c[v] = Out[(size_t)om * nCols + ocol];
        }
    }

#pragma unroll
    for (int kc = 0; kc < 4; ++kc) {
        int kb = kc * 32;
        float4 r0 = A4[(kb + kmod) >> 2];
        float4 r1 = A4[((kb + kmod) >> 2) + 1];
        float4 r2 = A4[(kb + kmod + 16) >> 2];
        float4 r3 = A4[((kb + kmod + 16) >> 2) + 1];
        v16bf a;
        a[0]  = (__bf16)(r0.x * inv); a[1]  = (__bf16)(r0.y * inv);
        a[2]  = (__bf16)(r0.z * inv); a[3]  = (__bf16)(r0.w * inv);
        a[4]  = (__bf16)(r1.x * inv); a[5]  = (__bf16)(r1.y * inv);
        a[6]  = (__bf16)(r1.z * inv); a[7]  = (__bf16)(r1.w * inv);
        a[8]  = (__bf16)(r2.x * inv); a[9]  = (__bf16)(r2.y * inv);
        a[10] = (__bf16)(r2.z * inv); a[11] = (__bf16)(r2.w * inv);
        a[12] = (__bf16)(r3.x * inv); a[13] = (__bf16)(r3.y * inv);
        a[14] = (__bf16)(r3.z * inv); a[15] = (__bf16)(r3.w * inv);
        v16bf b = Bv[(size_t)(wv * 4 + kc) * 32 + lane];
        c = __builtin_amdgcn_wmma_f32_16x16x32_bf16(false, a, false, b,
                                                    (short)0, c, false, false);
    }

    float bv = bias ? bias[ocol] : 0.0f;
    if (rowT + 16 <= nRows) {                       // uniform fast path: no predication
#pragma unroll
        for (int v = 0; v < 8; ++v)
            Out[(size_t)(rowT + hi8 + v) * nCols + ocol] = c[v] + bv;
    } else {
#pragma unroll
        for (int v = 0; v < 8; ++v) {
            int om = rowT + hi8 + v;
            if (om < nRows) Out[(size_t)om * nCols + ocol] = c[v] + bv;
        }
    }
}

__global__ void k_relu(float* __restrict__ p, int n) {
    for (int i = blockIdx.x * blockDim.x + threadIdx.x; i < n; i += gridDim.x * blockDim.x)
        p[i] = fmaxf(p[i], 0.0f);
}

// ---------- host orchestration ----------
extern "C" void kernel_launch(void* const* d_in, const int* in_sizes, int n_in,
                              void* d_out, int out_size, void* d_ws, size_t ws_size,
                              hipStream_t stream) {
    (void)in_sizes; (void)n_in; (void)out_size; (void)ws_size;
    static const int nodeCnt[10] = {50000,2000,80000,10000,2000,150000,120000,100000,60000,40000};
    static const int relSrc[26]  = {0,1,0,2,0,3,0,4,0,5,0,6,6,7,5,6,5,7,4,5,4,3,2,8,2,9};
    static const int relDst[26]  = {1,0,2,0,3,0,4,0,5,0,6,0,7,6,6,5,7,5,5,4,3,4,8,2,9,2};
    static const int relEi[26]   = {0,0,1,1,2,2,3,3,4,4,5,5,6,6,7,7,8,8,9,9,10,10,11,11,12,12};
    static const int relFlip[26] = {0,1,0,1,0,1,0,1,0,1,0,1,0,1,0,1,0,1,0,1,0,1,0,1,0,1};
    static const int needed0[10] = {1,1,1,1,1,1,1,0,0,0};   // types feeding layer-2 dst=course

    const float* Wl   = (const float*)d_in[23];
    const float* bl   = (const float*)d_in[24];
    const float* Wr   = (const float*)d_in[25];
    const float* linW = (const float*)d_in[26];
    const float* linb = (const float*)d_in[27];

    // workspace carve-up (256B aligned)
    char* ws = (char*)d_ws;
    auto align256 = [](size_t x) { return (x + 255) & ~(size_t)255; };
    size_t off = 0;
    float* xb[10] = {};
    for (int t = 0; t < 10; ++t) {
        if (!needed0[t]) continue;
        xb[t] = (float*)(ws + off);
        off = align256(off + (size_t)nodeCnt[t] * H * sizeof(float));
    }
    float* courseOut = (float*)(ws + off); off = align256(off + (size_t)50000 * H * sizeof(float));
    float* agg       = (float*)(ws + off); off = align256(off + (size_t)150000 * H * sizeof(float));
    float* deg       = (float*)(ws + off); off = align256(off + (size_t)150000 * sizeof(float));
    __bf16* Wlbf     = (__bf16*)(ws + off); off = align256(off + (size_t)2 * 26 * H * H * sizeof(__bf16));
    __bf16* WrSumbf  = (__bf16*)(ws + off); off = align256(off + (size_t)2 * 10 * H * H * sizeof(__bf16));
    float* bSum      = (float*)(ws + off);  off = align256(off + (size_t)2 * 10 * H * sizeof(float));
    __bf16* linWbf   = (__bf16*)(ws + off); off = align256(off + (size_t)H * OUTC * sizeof(__bf16));

    // ---- prep: weight conversion (swizzled) / presums ----
    {
        int n = 2 * 26 * H * H;
        k_cvt_swz<<<(n + 255) / 256, 256, 0, stream>>>(Wl, Wlbf, 2 * 26, H);
        int m = 2 * 10 * H * H;
        k_wrsum<<<(m + 255) / 256, 256, 0, stream>>>(Wr, WrSumbf, bl, bSum);
        int p = H * OUTC;
        k_cvt_swz<<<(p + 255) / 256, 256, 0, stream>>>(linW, linWbf, 1, OUTC);
    }

    // ---- layer loop ----
    for (int layer = 0; layer < 2; ++layer) {
        const float* xin[10];
        for (int t = 0; t < 10; ++t)
            xin[t] = (layer == 0) ? (const float*)d_in[t] : (const float*)xb[t];

        int need[10];
        for (int t = 0; t < 10; ++t) need[t] = (layer == 0) ? needed0[t] : (t == 0);

        // self GEMM (writes out = x @ WrSum + bSum)
        for (int t = 0; t < 10; ++t) {
            if (!need[t]) continue;
            float* outT = (layer == 0) ? xb[t] : courseOut;
            int n = nodeCnt[t];
            k_gemm<<<(n + 15) / 16, 2 * H, 0, stream>>>(
                xin[t], nullptr, WrSumbf + (size_t)(layer * 10 + t) * H * H,
                bSum + (size_t)(layer * 10 + t) * H, outT, n, H, 0);
        }

        // relation passes (accumulate out += mean @ Wl)
        for (int r = 0; r < 26; ++r) {
            int d = relDst[r];
            if (!need[d]) continue;
            int s = relSrc[r];
            int nd = nodeCnt[d];
            hipMemsetAsync(agg, 0, (size_t)nd * H * sizeof(float), stream);
            hipMemsetAsync(deg, 0, (size_t)nd * sizeof(float), stream);
            const int* ei = (const int*)d_in[10 + relEi[r]];
            const int* srcI = ei + (relFlip[r] ? NE : 0);
            const int* dstI = ei + (relFlip[r] ? 0 : NE);
            k_scatter<<<(NE + 7) / 8, 256, 0, stream>>>(xin[s], srcI, dstI, agg, deg, NE);
            float* outT = (layer == 0) ? xb[d] : courseOut;
            k_gemm<<<(nd + 15) / 16, 2 * H, 0, stream>>>(
                agg, deg, Wlbf + (size_t)(layer * 26 + r) * H * H,
                nullptr, outT, nd, H, 1);
        }

        // relu
        if (layer == 0) {
            for (int t = 0; t < 10; ++t) {
                if (!need[t]) continue;
                k_relu<<<1024, 256, 0, stream>>>(xb[t], nodeCnt[t] * H);
            }
        } else {
            k_relu<<<1024, 256, 0, stream>>>(courseOut, 50000 * H);
        }
    }

    // ---- final projection: d_out = relu(courseOut) @ lin_W + lin_b ----
    k_gemm<<<(50000 + 15) / 16, 2 * OUTC, 0, stream>>>(
        courseOut, nullptr, linWbf, linb, (float*)d_out, 50000, OUTC, 0);
}
